// MessagePassing_87505663689045
// MI455X (gfx1250) — matrix operand order repaired and checked by reading, not compiled, not run on hardware
//
#include <hip/hip_runtime.h>
#include <hip/hip_bf16.h>
#include <math.h>
#include <stdint.h>

// Problem constants (B=4, At=256, Nbr=16, F=128)
#define F_      128
#define TWO_F   256
#define AT_     256
#define NBR_    16
#define NK_     15
#define R1_     16384        // B*At*Nbr
#define R3_     245760       // R1*Nk
#define EPS_    1e-5f

typedef __attribute__((ext_vector_type(16))) _Float16 v16h;
typedef __attribute__((ext_vector_type(8)))  _Float16 v8h;
typedef __attribute__((ext_vector_type(8)))  float    v8f;

__device__ __forceinline__ float sigmoidf_(float x) { return 1.0f / (1.0f + __expf(-x)); }

// Low 32 bits of a flat pointer to a __shared__ object == wave-relative LDS
// byte offset (LDS aperture base has zero low bits; ISA 10.2 aperture rules).
__device__ __forceinline__ uint32_t lds_off(const void* p) {
  return (uint32_t)(uintptr_t)p;
}

// CDNA5 async global->LDS DMA (tracked by ASYNCcnt). One 16B chunk per lane.
__device__ __forceinline__ void async_g2l_b128(uint32_t lds_byte_off, const void* gsrc) {
  asm volatile("global_load_async_to_lds_b128 %0, %1, off"
               :: "v"(lds_byte_off), "v"(gsrc) : "memory");
}
__device__ __forceinline__ void wait_async0() {
  asm volatile("s_wait_asynccnt 0" ::: "memory");
}

// ---- WMMA fragment loaders (wave32, contiguous k order in A and B) -------------------
// A: 16x32 f16 tile, row-major in LDS, ld=32. Per lane: 16 contiguous halves
// (k = hi*16 + j) -> compiler emits ds_load_b128 x2.
__device__ __forceinline__ v16h load_a_frag(const _Float16* t, int lane) {
  const int row = lane & 15, hi = lane >> 4;
  v16h a;
#pragma unroll
  for (int j = 0; j < 16; ++j) {
    int k = (hi << 4) + j;
    a[j] = t[row * 32 + k];
  }
  return a;
}
// B: 32x16 (K x N) fragment loaded straight from global memory, where the
// weight matrix is pre-transposed to column-major f16 (Wt[n*K + k]).
// Lane holds column n = lane&15; k = (lane>>4)*16 + j, contiguous -> b128 x2.
__device__ __forceinline__ v16h load_b_frag_g(const _Float16* p) {
  v8h b0 = *(const v8h*)p;
  v8h b1 = *(const v8h*)(p + 8);
  v16h b;
#pragma unroll
  for (int j = 0; j < 8; ++j) { b[j] = b0[j]; b[j + 8] = b1[j]; }
  return b;
}

// ---------------------------------------------------------------------------
// Prep: weight transpose+convert  Wt[n*K+k] = (f16) W[k*N+n], N=256
// ---------------------------------------------------------------------------
__global__ void wtrans_kernel(const float* __restrict__ W, int K, _Float16* __restrict__ Wt) {
  int i = blockIdx.x * 256 + threadIdx.x;   // over K*256
  if (i < K * TWO_F) {
    int k = i >> 8, n = i & 255;
    Wt[n * K + k] = (_Float16)W[i];
  }
}

// Prep: C1h[r, 0:256] = mask * concat(node_i, edge);  EDGEh = f16 copy of edge
__global__ __launch_bounds__(256) void prep_c1_kernel(
    const float* __restrict__ node, const float* __restrict__ edge,
    const int* __restrict__ mask, _Float16* __restrict__ C1h, _Float16* __restrict__ EDGEh) {
  const int r = blockIdx.x, c = threadIdx.x;
  const int a = (r >> 4) & 255, bb = r >> 12;
  const float mf = (mask[r] != 0) ? 1.0f : 0.0f;
  float v = (c < F_) ? node[(bb * AT_ + a) * F_ + c] : edge[r * F_ + c - F_];
  C1h[(size_t)r * TWO_F + c] = (_Float16)(v * mf);
  if (c < F_) EDGEh[r * F_ + c] = (_Float16)edge[r * F_ + c];
}

// Prep: C2h[r, c] = mask * un_i * un_j  (f16)
__global__ __launch_bounds__(128) void prep_c2_kernel(
    const float* __restrict__ un, const int* __restrict__ idx,
    const int* __restrict__ mask, _Float16* __restrict__ C2h) {
  const int r = blockIdx.x, c = threadIdx.x;
  const int a = (r >> 4) & 255, bb = r >> 12;
  const float mf = (mask[r] != 0) ? 1.0f : 0.0f;
  float v = un[(bb * AT_ + a) * F_ + c] * un[(bb * AT_ + idx[r]) * F_ + c] * mf;
  C2h[(size_t)r * F_ + c] = (_Float16)v;
}

// ---------------------------------------------------------------------------
// Generic GEMM: Y[M x 256] = A[M x K](f16, row-major) @ Wt(f16, col-major)
// A tiles DMA'd to LDS with async b128; B frags loaded global->VGPR.
// Serves stage 1 (C1h, K=256) and stage 2 (C2h, K=128).
// ---------------------------------------------------------------------------
__global__ __launch_bounds__(256) void gemm_f16_kernel(
    const _Float16* __restrict__ A, const _Float16* __restrict__ Wt,
    int K, float* __restrict__ Y) {
  __shared__ _Float16 As[64 * 32];
  const int tid = threadIdx.x;
  const int m0 = blockIdx.x * 64, n0 = blockIdx.y * 32;
  const int wave = tid >> 5, lane = tid & 31;
  const int wm = wave >> 1, wn = wave & 1;
  const int row = tid >> 2, chunk = tid & 3;           // 64 rows x 4 x 16B
  const uint32_t aoff = lds_off(As) + (uint32_t)(row * 64 + chunk * 16);
  const _Float16* arow = A + (size_t)(m0 + row) * K + chunk * 8;
  const _Float16* wp = Wt + (size_t)(n0 + wn * 16 + (lane & 15)) * K + (lane >> 4) * 16;
  v8f acc = {};
  for (int kk = 0; kk < K; kk += 32) {
    async_g2l_b128(aoff, arow + kk);
    wait_async0();
    __syncthreads();
    v16h a = load_a_frag(&As[wm * 16 * 32], lane);
    v16h b = load_b_frag_g(wp + kk);
    acc = __builtin_amdgcn_wmma_f32_16x16x32_f16(false, a, false, b, (short)0, acc, false, false);
    __syncthreads();
  }
  const int cg = n0 + wn * 16 + (lane & 15);
  const int hi = lane >> 4;
#pragma unroll
  for (int vg = 0; vg < 8; ++vg) {
    int rg = m0 + wm * 16 + vg + 8 * hi;
    Y[rg * TWO_F + cg] = acc[vg];
  }
}

// ---- 3-body row metadata (offsets in f16 elements) ------------------------
__device__ __forceinline__ void c3_rowinfo(int r3, const int* __restrict__ idx,
                                           int& ui, int& uj, int& uk, int& eij, int& ekj) {
  int r1 = r3 / NK_, kp = r3 - r1 * NK_;
  int n = r1 & 15, a = (r1 >> 4) & 255, bb = r1 >> 12;
  int nk = kp + (kp >= n ? 1 : 0);
  int base = (r1 & ~15) | nk;
  ui  = (bb * AT_ + a) * F_;
  uj  = (bb * AT_ + idx[r1]) * F_;
  uk  = (bb * AT_ + idx[base]) * F_;
  eij = r1 * F_;
  ekj = base * F_;
}

// ---------------------------------------------------------------------------
// GEMM3 pass A: Y3 = c3 @ W3 (M=245760, K=640, N=256); only column stats.
// A tile = per-row gathered async DMA from UNh/EDGEh (segment constant per
// 32-wide K step since 32 | 128).
// ---------------------------------------------------------------------------
__global__ __launch_bounds__(256) void gemm3_stats_kernel(
    const _Float16* __restrict__ UNh, const _Float16* __restrict__ EDGEh,
    const int* __restrict__ idx, const _Float16* __restrict__ Wt3,
    float* __restrict__ s3raw) {
  __shared__ _Float16 As[64 * 32];
  __shared__ int ri[5][64];
  const int tid = threadIdx.x;
  const int m0 = blockIdx.x * 64, n0 = blockIdx.y * 32;
  if (tid < 64) {
    int ui, uj, uk, eij, ekj;
    c3_rowinfo(m0 + tid, idx, ui, uj, uk, eij, ekj);
    ri[0][tid] = ui; ri[1][tid] = uj; ri[2][tid] = uk; ri[3][tid] = eij; ri[4][tid] = ekj;
  }
  __syncthreads();
  const int wave = tid >> 5, lane = tid & 31;
  const int wm = wave >> 1, wn = wave & 1;
  const int row = tid >> 2, chunk = tid & 3;
  const uint32_t aoff = lds_off(As) + (uint32_t)(row * 64 + chunk * 16);
  const _Float16* wp = Wt3 + (size_t)(n0 + wn * 16 + (lane & 15)) * (5 * F_) + (lane >> 4) * 16;
  v8f acc = {};
  for (int kk = 0; kk < 5 * F_; kk += 32) {
    const int seg = kk >> 7, o0 = (kk & 127) + chunk * 8;
    const _Float16* src = ((seg < 3) ? UNh : EDGEh) + ri[seg][row] + o0;
    async_g2l_b128(aoff, src);
    wait_async0();
    __syncthreads();
    v16h a = load_a_frag(&As[wm * 16 * 32], lane);
    v16h b = load_b_frag_g(wp + kk);
    acc = __builtin_amdgcn_wmma_f32_16x16x32_f16(false, a, false, b, (short)0, acc, false, false);
    __syncthreads();
  }
  const int cg = n0 + wn * 16 + (lane & 15);
  float s = 0.f, q = 0.f;
#pragma unroll
  for (int vg = 0; vg < 8; ++vg) { s += acc[vg]; q += acc[vg] * acc[vg]; }
  atomicAdd(&s3raw[cg], s);
  atomicAdd(&s3raw[TWO_F + cg], q);
}

// ---------------------------------------------------------------------------
// GEMM3 pass B: recompute Y3 for paired columns (c, c+128), BN3 +
// sigmoid(g)*tanh(e), atomic reduce over k' (Nk=15) into TB3[r1,c].
// ---------------------------------------------------------------------------
__global__ __launch_bounds__(256) void gemm3_apply_kernel(
    const _Float16* __restrict__ UNh, const _Float16* __restrict__ EDGEh,
    const int* __restrict__ idx, const _Float16* __restrict__ Wt3,
    const float* __restrict__ mu3, const float* __restrict__ rs3,
    const float* __restrict__ g3, const float* __restrict__ be3,
    float* __restrict__ TB3) {
  __shared__ _Float16 As[64 * 32];
  __shared__ int ri[5][64];
  const int tid = threadIdx.x;
  const int m0 = blockIdx.x * 64, n0 = blockIdx.y * 32;   // n0 in [0,128)
  if (tid < 64) {
    int ui, uj, uk, eij, ekj;
    c3_rowinfo(m0 + tid, idx, ui, uj, uk, eij, ekj);
    ri[0][tid] = ui; ri[1][tid] = uj; ri[2][tid] = uk; ri[3][tid] = eij; ri[4][tid] = ekj;
  }
  __syncthreads();
  const int wave = tid >> 5, lane = tid & 31;
  const int wm = wave >> 1, wn = wave & 1;
  const int row = tid >> 2, chunk = tid & 3;
  const uint32_t aoff = lds_off(As) + (uint32_t)(row * 64 + chunk * 16);
  const int cg = n0 + wn * 16 + (lane & 15);
  const _Float16* wpg = Wt3 + (size_t)cg * (5 * F_) + (lane >> 4) * 16;
  const _Float16* wpe = Wt3 + (size_t)(cg + F_) * (5 * F_) + (lane >> 4) * 16;
  v8f accg = {}, acce = {};
  for (int kk = 0; kk < 5 * F_; kk += 32) {
    const int seg = kk >> 7, o0 = (kk & 127) + chunk * 8;
    const _Float16* src = ((seg < 3) ? UNh : EDGEh) + ri[seg][row] + o0;
    async_g2l_b128(aoff, src);
    wait_async0();
    __syncthreads();
    v16h a  = load_a_frag(&As[wm * 16 * 32], lane);
    v16h bg = load_b_frag_g(wpg + kk);
    v16h be = load_b_frag_g(wpe + kk);
    accg = __builtin_amdgcn_wmma_f32_16x16x32_f16(false, a, false, bg, (short)0, accg, false, false);
    acce = __builtin_amdgcn_wmma_f32_16x16x32_f16(false, a, false, be, (short)0, acce, false, false);
    __syncthreads();
  }
  const int hi = lane >> 4;
  const float mg = mu3[cg],      rg = rs3[cg],      gg = g3[cg],      bg_ = be3[cg];
  const float me = mu3[cg + F_], re = rs3[cg + F_], ge = g3[cg + F_], be_ = be3[cg + F_];
#pragma unroll
  for (int vg = 0; vg < 8; ++vg) {
    int r3 = m0 + wm * 16 + vg + 8 * hi;
    float gv = (accg[vg] - mg) * rg * gg + bg_;
    float ev = (acce[vg] - me) * re * ge + be_;
    float h = sigmoidf_(gv) * tanhf(ev);
    int r1 = r3 / NK_;
    atomicAdd(&TB3[r1 * F_ + cg], h);
  }
}

// ---- column mean / inv-std over [rows x cols] -----------------------------
__global__ __launch_bounds__(256) void colstats_kernel(
    const float* __restrict__ X, int rows, int cols,
    float* __restrict__ mu, float* __restrict__ rs) {
  __shared__ float s1[256], s2[256];
  const int c = blockIdx.x, t = threadIdx.x;
  float a = 0.f, b = 0.f;
  for (int r = t; r < rows; r += 256) {
    float v = X[r * cols + c];
    a += v; b += v * v;
  }
  s1[t] = a; s2[t] = b;
  __syncthreads();
  for (int off = 128; off; off >>= 1) {
    if (t < off) { s1[t] += s1[t + off]; s2[t] += s2[t + off]; }
    __syncthreads();
  }
  if (t == 0) {
    float m = s1[0] / (float)rows;
    mu[c] = m;
    rs[c] = rsqrtf(s2[0] / (float)rows - m * m + EPS_);
  }
}

__global__ void finalize_stats3_kernel(const float* __restrict__ s3raw,
                                       float* __restrict__ mu3, float* __restrict__ rs3) {
  int c = threadIdx.x;  // 256 threads
  float m = s3raw[c] / (float)R3_;
  mu3[c] = m;
  rs3[c] = rsqrtf(s3raw[TWO_F + c] / (float)R3_ - m * m + EPS_);
}

__global__ void zero_kernel(float* __restrict__ p, int n) {
  int i = blockIdx.x * 256 + threadIdx.x;
  if (i < n) p[i] = 0.f;
}

// nbr_sumed: sum over Nbr of sigmoid(bn(g)) * tanh(bn(e))
__global__ __launch_bounds__(128) void act1_rowsum_kernel(
    const float* __restrict__ Y, const float* __restrict__ mu1, const float* __restrict__ rs1,
    const float* __restrict__ g1, const float* __restrict__ be1, float* __restrict__ S) {
  const int c = threadIdx.x, ba = blockIdx.x;
  const float mg = mu1[c],      rg = rs1[c],      gg = g1[c],      bg = be1[c];
  const float me = mu1[c + F_], re = rs1[c + F_], ge = g1[c + F_], be = be1[c + F_];
  float acc = 0.f;
  for (int n = 0; n < NBR_; ++n) {
    int r = ba * NBR_ + n;
    float yg = Y[r * TWO_F + c];
    float ye = Y[r * TWO_F + F_ + c];
    acc += sigmoidf_((yg - mg) * rg * gg + bg) * tanhf((ye - me) * re * ge + be);
  }
  S[ba * F_ + c] = acc;
}

// un = tanh(node + bn2(S)); write f32 (ws + d_out) and f16 (for gemm3 gathers)
__global__ __launch_bounds__(128) void finish_un_kernel(
    const float* __restrict__ node, const float* __restrict__ S,
    const float* __restrict__ mu2, const float* __restrict__ rs2,
    const float* __restrict__ g2, const float* __restrict__ be2,
    float* __restrict__ un_ws, _Float16* __restrict__ UNh, float* __restrict__ out) {
  const int c = threadIdx.x;
  const int i = blockIdx.x * F_ + c;
  float v = tanhf(node[i] + (S[i] - mu2[c]) * rs2[c] * g2[c] + be2[c]);
  un_ws[i] = v;
  UNh[i] = (_Float16)v;
  out[i] = v;
}

// two_body = sigmoid(bn(g2)) * tanh(bn(e2))
__global__ __launch_bounds__(128) void two_body_kernel(
    const float* __restrict__ Y, const float* __restrict__ mu, const float* __restrict__ rs,
    const float* __restrict__ g, const float* __restrict__ be, float* __restrict__ TB) {
  const int c = threadIdx.x, r = blockIdx.x;
  float yg = Y[r * TWO_F + c];
  float ye = Y[r * TWO_F + F_ + c];
  float gv = (yg - mu[c]) * rs[c] * g[c] + be[c];
  float ev = (ye - mu[c + F_]) * rs[c + F_] * g[c + F_] + be[c + F_];
  TB[r * F_ + c] = sigmoidf_(gv) * tanhf(ev);
}

// ue = tanh(edge + two_body + bn_sum(TB3))
__global__ __launch_bounds__(128) void final_ue_kernel(
    const float* __restrict__ edge, const float* __restrict__ TB, const float* __restrict__ TB3,
    const float* __restrict__ muS, const float* __restrict__ rsS,
    const float* __restrict__ gS, const float* __restrict__ beS, float* __restrict__ out_ue) {
  const int c = threadIdx.x;
  const int i = blockIdx.x * F_ + c;
  float t3 = (TB3[i] - muS[c]) * rsS[c] * gS[c] + beS[c];
  out_ue[i] = tanhf(edge[i] + TB[i] + t3);
}

// ---- workspace layout ------------------------------------------------------
// float region
static constexpr size_t OFF_Y    = 0;                              // 16384*256 (reused)
static constexpr size_t OFF_S    = OFF_Y    + (size_t)R1_ * TWO_F;
static constexpr size_t OFF_UN   = OFF_S    + 1024 * F_;
static constexpr size_t OFF_TB   = OFF_UN   + 1024 * F_;
static constexpr size_t OFF_TB3  = OFF_TB   + (size_t)R1_ * F_;
static constexpr size_t OFF_MU1  = OFF_TB3  + (size_t)R1_ * F_;
static constexpr size_t OFF_RS1  = OFF_MU1  + 256;
static constexpr size_t OFF_MU2  = OFF_RS1  + 256;
static constexpr size_t OFF_RS2  = OFF_MU2  + 128;
static constexpr size_t OFF_MU2B = OFF_RS2  + 128;
static constexpr size_t OFF_RS2B = OFF_MU2B + 256;
static constexpr size_t OFF_S3R  = OFF_RS2B + 256;                 // 512
static constexpr size_t OFF_MU3  = OFF_S3R  + 512;
static constexpr size_t OFF_RS3  = OFF_MU3  + 256;
static constexpr size_t OFF_MUS  = OFF_RS3  + 256;
static constexpr size_t OFF_RSS  = OFF_MUS  + 128;
static constexpr size_t OFF_HALF = OFF_RSS  + 128;                 // start of f16 region (floats)
// half region (offsets in _Float16 elements from half base)
static constexpr size_t H_WTN  = 0;                                //  256*256
static constexpr size_t H_WT2  = H_WTN  + 65536;                   //  256*128
static constexpr size_t H_WT3  = H_WT2  + 32768;                   //  256*640
static constexpr size_t H_C1   = H_WT3  + 163840;                  //  R1*256
static constexpr size_t H_C2   = H_C1   + (size_t)R1_ * TWO_F;     //  R1*128
static constexpr size_t H_EDGE = H_C2   + (size_t)R1_ * F_;        //  R1*128
static constexpr size_t H_UN   = H_EDGE + (size_t)R1_ * F_;        //  1024*128

extern "C" void kernel_launch(void* const* d_in, const int* in_sizes, int n_in,
                              void* d_out, int out_size, void* d_ws, size_t ws_size,
                              hipStream_t stream) {
  (void)in_sizes; (void)n_in; (void)out_size; (void)ws_size;
  const float* node = (const float*)d_in[0];
  const float* edge = (const float*)d_in[1];
  const int*   idx  = (const int*)d_in[2];
  const int*   mask = (const int*)d_in[3];
  const float* Wn   = (const float*)d_in[4];
  const float* g1   = (const float*)d_in[6];
  const float* be1  = (const float*)d_in[7];
  const float* g2   = (const float*)d_in[8];
  const float* be2  = (const float*)d_in[9];
  const float* W2   = (const float*)d_in[10];
  const float* g2b  = (const float*)d_in[12];
  const float* be2b = (const float*)d_in[13];
  const float* W3   = (const float*)d_in[14];
  const float* g3b  = (const float*)d_in[16];
  const float* be3b = (const float*)d_in[17];
  const float* gS   = (const float*)d_in[18];
  const float* beS  = (const float*)d_in[19];
  // biases b_node/b2/b3 (indices 5,11,15) cancel inside batch-norm.

  float* out = (float*)d_out;
  float* ws  = (float*)d_ws;
  float* Y    = ws + OFF_Y;
  float* S    = ws + OFF_S;
  float* UN   = ws + OFF_UN;
  float* TB   = ws + OFF_TB;
  float* TB3  = ws + OFF_TB3;
  float* mu1  = ws + OFF_MU1;   float* rs1  = ws + OFF_RS1;
  float* mu2  = ws + OFF_MU2;   float* rs2  = ws + OFF_RS2;
  float* mu2b = ws + OFF_MU2B;  float* rs2b = ws + OFF_RS2B;
  float* s3r  = ws + OFF_S3R;
  float* mu3  = ws + OFF_MU3;   float* rs3  = ws + OFF_RS3;
  float* muS  = ws + OFF_MUS;   float* rsS  = ws + OFF_RSS;
  _Float16* hb   = (_Float16*)(ws + OFF_HALF);
  _Float16* WtN  = hb + H_WTN;
  _Float16* Wt2  = hb + H_WT2;
  _Float16* Wt3  = hb + H_WT3;
  _Float16* C1h  = hb + H_C1;
  _Float16* C2h  = hb + H_C2;
  _Float16* Eh   = hb + H_EDGE;
  _Float16* UNh  = hb + H_UN;

  // Prep: f16 transposed weights + stage-1 A operand
  wtrans_kernel<<<(256 * TWO_F + 255) / 256, 256, 0, stream>>>(Wn, TWO_F, WtN);
  wtrans_kernel<<<(128 * TWO_F + 255) / 256, 256, 0, stream>>>(W2, F_, Wt2);
  wtrans_kernel<<<(640 * TWO_F + 255) / 256, 256, 0, stream>>>(W3, 5 * F_, Wt3);
  prep_c1_kernel<<<R1_, 256, 0, stream>>>(node, edge, mask, C1h, Eh);

  // Stage 1: first body GEMM + BN1 + gated neighbor sum -> un
  gemm_f16_kernel<<<dim3(R1_ / 64, TWO_F / 32), 256, 0, stream>>>(C1h, WtN, TWO_F, Y);
  colstats_kernel<<<TWO_F, 256, 0, stream>>>(Y, R1_, TWO_F, mu1, rs1);
  act1_rowsum_kernel<<<1024, 128, 0, stream>>>(Y, mu1, rs1, g1, be1, S);
  colstats_kernel<<<F_, 256, 0, stream>>>(S, 1024, F_, mu2, rs2);
  finish_un_kernel<<<1024, 128, 0, stream>>>(node, S, mu2, rs2, g2, be2, UN, UNh, out);

  // Stage 2: two-body GEMM (reuses Y) + BN2b + gate
  prep_c2_kernel<<<R1_, 128, 0, stream>>>(UN, idx, mask, C2h);
  gemm_f16_kernel<<<dim3(R1_ / 64, TWO_F / 32), 256, 0, stream>>>(C2h, Wt2, F_, Y);
  colstats_kernel<<<TWO_F, 256, 0, stream>>>(Y, R1_, TWO_F, mu2b, rs2b);
  two_body_kernel<<<R1_, 128, 0, stream>>>(Y, mu2b, rs2b, g2b, be2b, TB);

  // Stage 3: three-body GEMM, two passes (stats, then apply+reduce over Nk)
  zero_kernel<<<2, 256, 0, stream>>>(s3r, 512);
  zero_kernel<<<(R1_ * F_ + 255) / 256, 256, 0, stream>>>(TB3, R1_ * F_);
  gemm3_stats_kernel<<<dim3(R3_ / 64, TWO_F / 32), 256, 0, stream>>>(UNh, Eh, idx, Wt3, s3r);
  finalize_stats3_kernel<<<1, 256, 0, stream>>>(s3r, mu3, rs3);
  gemm3_apply_kernel<<<dim3(R3_ / 64, F_ / 32), 256, 0, stream>>>(
      UNh, Eh, idx, Wt3, mu3, rs3, g3b, be3b, TB3);

  // Stage 4: BN over summed three-body, final edge update
  colstats_kernel<<<F_, 256, 0, stream>>>(TB3, R1_, F_, muS, rsS);
  final_ue_kernel<<<R1_, 128, 0, stream>>>(edge, TB, TB3, muS, rsS, gS, beS,
                                           out + (size_t)1024 * F_);
}
